// GGNN_17575006175918
// MI455X (gfx1250) — matrix operand order repaired and checked by reading, not compiled, not run on hardware
//
#include <hip/hip_runtime.h>
#include <hip/hip_bf16.h>
#include <cstdint>

// ---------------------------------------------------------------------------
// Types for WMMA (CDNA5 / gfx1250, wave32)
// ---------------------------------------------------------------------------
typedef __attribute__((ext_vector_type(16))) __bf16 v16bf;
typedef __attribute__((ext_vector_type(8)))  __bf16 v8bf;
typedef __attribute__((ext_vector_type(4)))  __bf16 v4bf;
typedef __attribute__((ext_vector_type(8)))  float  v8f;

#define TILE_M 128
#define TILE_N 128
#define TILE_K 32
#define LDS_STRIDE 40   // bf16 elements; 80 bytes -> 16B aligned rows

#define FLAG_ACC  1
#define FLAG_RELU 2

// ---------------------------------------------------------------------------
// WMMA GEMM: C[M,N] (+)= A[M,K] @ B[K,N] (+ bias[N]) (+ relu)
// A row-major (lda=K, rows readable up to Mpad), B row stride ldb with columns
// readable up to Npad (>=N; pad values are garbage and only feed masked
// outputs), C row stride ldc with column offset coff.
// fp32 in memory, bf16 matrix cores, fp32 accumulate.
// Block tile 128x128, 8 waves, each wave 32(M)x64(N) = 8 WMMA per K-chunk.
// Double-buffered LDS: stage chunk i+1 while computing chunk i; one barrier/iter.
// ---------------------------------------------------------------------------
__global__ __launch_bounds__(256) void gemm_nn_bf16(
    const float* __restrict__ A, const float* __restrict__ B,
    const float* __restrict__ bias, float* __restrict__ C,
    int M, int N, int K, int Mpad, int ldb, int Npad,
    int ldc, int coff, int flags)
{
    alignas(16) __shared__ __bf16 sA[2][TILE_M][LDS_STRIDE];   // [buf][row][k]
    alignas(16) __shared__ __bf16 sBT[2][TILE_N][LDS_STRIDE];  // [buf][col][k]

    const int tid  = threadIdx.x;
    const int lane = tid & 31;
    const int wave = tid >> 5;
    const int wm   = wave & 3;   // 4 waves across M (32 rows each)
    const int wn   = wave >> 2;  // 2 waves across N (64 cols each)
    const int row0 = blockIdx.y * TILE_M;
    const int col0 = blockIdx.x * TILE_N;

    const int half = lane >> 4;  // 0: lanes 0-15, 1: lanes 16-31
    const int l15  = lane & 15;
    const int kbA  = half * 8;   // A-fragment K base per ISA layout

    const bool aVec = (row0 + TILE_M <= Mpad) && ((K & 3) == 0);
    const bool bVec = (col0 + TILE_N <= Npad) && ((ldb & 3) == 0);

    // ---- staging helpers ----
    auto stageA = [&](int buf, int k0) {
        if (aVec && (k0 + TILE_K <= K)) {
            // 4096 floats / 256 threads = 4 float4 per thread, coalesced in K.
            #pragma unroll
            for (int i = 0; i < 4; ++i) {
                int lin4 = tid + i * 256;
                int r  = lin4 >> 3;          // 8 float4 per 32-wide row
                int kk = (lin4 & 7) * 4;
                const float4 f = *(const float4*)(A + (size_t)(row0 + r) * K + k0 + kk);
                v4bf p; p[0] = (__bf16)f.x; p[1] = (__bf16)f.y;
                        p[2] = (__bf16)f.z; p[3] = (__bf16)f.w;
                *(v4bf*)&sA[buf][r][kk] = p;
            }
        } else {
            // phase 1: batched clamped loads (no divergence, single wait)
            float va[16];
            #pragma unroll
            for (int i = 0; i < 16; ++i) {
                int lin = tid + i * 256;
                int gr = row0 + (lin >> 5), gk = k0 + (lin & 31);
                int cr = gr < M ? gr : M - 1;
                int ck = gk < K ? gk : K - 1;
                va[i] = A[(size_t)cr * K + ck];
            }
            // phase 2: select + convert + store
            #pragma unroll
            for (int i = 0; i < 16; ++i) {
                int lin = tid + i * 256;
                int r = lin >> 5, kk = lin & 31;
                bool ok = (row0 + r < M) && (k0 + kk < K);
                sA[buf][r][kk] = (__bf16)(ok ? va[i] : 0.0f);
            }
        }
    };
    auto stageB = [&](int buf, int k0) {
        if (bVec && (k0 + TILE_K <= K)) {
            #pragma unroll
            for (int i = 0; i < 4; ++i) {
                int lin4 = tid + i * 256;
                int r = lin4 >> 5;           // 32 float4 per 128-wide row
                int c = (lin4 & 31) * 4;
                const float4 f = *(const float4*)(B + (size_t)(k0 + r) * ldb + col0 + c);
                sBT[buf][c + 0][r] = (__bf16)f.x;
                sBT[buf][c + 1][r] = (__bf16)f.y;
                sBT[buf][c + 2][r] = (__bf16)f.z;
                sBT[buf][c + 3][r] = (__bf16)f.w;
            }
        } else {
            float vb[16];
            #pragma unroll
            for (int i = 0; i < 16; ++i) {
                int lin = tid + i * 256;
                int gk = k0 + (lin >> 7), gc = col0 + (lin & 127);
                int ck = gk < K ? gk : K - 1;
                int cc = gc < N ? gc : N - 1;
                vb[i] = B[(size_t)ck * ldb + cc];
            }
            #pragma unroll
            for (int i = 0; i < 16; ++i) {
                int lin = tid + i * 256;
                int r = lin >> 7, c = lin & 127;
                bool ok = (k0 + r < K) && (col0 + c < N);
                sBT[buf][c][r] = (__bf16)(ok ? vb[i] : 0.0f);
            }
        }
    };

    v8f acc[2][4];
    #pragma unroll
    for (int i = 0; i < 2; ++i)
        #pragma unroll
        for (int j = 0; j < 4; ++j)
            #pragma unroll
            for (int v = 0; v < 8; ++v) acc[i][j][v] = 0.0f;

    const int nIter = (K + TILE_K - 1) / TILE_K;

    // prologue: fill buffer 0
    stageA(0, 0);
    stageB(0, 0);
    __syncthreads();

    for (int it = 0; it < nIter; ++it) {
        const int cur = it & 1;

        // stage next chunk into the other buffer while computing this one
        if (it + 1 < nIter) {
            stageA(cur ^ 1, (it + 1) * TILE_K);
            stageB(cur ^ 1, (it + 1) * TILE_K);
        }

        // ---- fragments: contiguous 16B chunks -> ds_load_b128 ----
        v16bf af[2], bfm[4];
        #pragma unroll
        for (int mi = 0; mi < 2; ++mi) {
            int r = wm * 32 + mi * 16 + l15;
            v8bf lo = *(const v8bf*)&sA[cur][r][kbA];       // K = kbA..kbA+7
            v8bf hi = *(const v8bf*)&sA[cur][r][16 + kbA];  // K = 16+kbA..+7
            af[mi] = __builtin_shufflevector(lo, hi, 0,1,2,3,4,5,6,7,
                                                     8,9,10,11,12,13,14,15);
        }
        #pragma unroll
        for (int ni = 0; ni < 4; ++ni) {
            int c = wn * 64 + ni * 16 + l15;
            v8bf lo = *(const v8bf*)&sBT[cur][c][half * 8];       // K = half*8..+7
            v8bf hi = *(const v8bf*)&sBT[cur][c][16 + half * 8];  // K = 16+half*8..+7
            bfm[ni] = __builtin_shufflevector(lo, hi, 0,1,2,3,4,5,6,7,
                                                      8,9,10,11,12,13,14,15);
        }

        #pragma unroll
        for (int mi = 0; mi < 2; ++mi)
            #pragma unroll
            for (int ni = 0; ni < 4; ++ni)
                acc[mi][ni] = __builtin_amdgcn_wmma_f32_16x16x32_bf16(
                    false, af[mi], false, bfm[ni], (short)0, acc[mi][ni],
                    false, false);

        // single barrier per iteration: next buffer staged, current fully read
        __syncthreads();
    }

    // Epilogue: C/D layout — VGPR v: M = v (lanes 0-15) / 8+v (lanes 16-31), N = lane&15
    #pragma unroll
    for (int ni = 0; ni < 4; ++ni) {
        const int col = col0 + wn * 64 + ni * 16 + l15;
        const bool cok = col < N;
        #pragma unroll
        for (int mi = 0; mi < 2; ++mi)
            #pragma unroll
            for (int v = 0; v < 8; ++v) {
                int row = row0 + wm * 32 + mi * 16 + half * 8 + v;
                if (cok && row < M) {
                    size_t idx = (size_t)row * ldc + coff + col;
                    float c = acc[mi][ni][v];
                    if (flags & FLAG_ACC) c += C[idx];
                    if (bias) c += bias[col];
                    if (flags & FLAG_RELU) c = fmaxf(c, 0.0f);
                    C[idx] = c;
                }
            }
    }
}

// ---------------------------------------------------------------------------
// Small transposed GEMM with split-K atomics: C[M,N] += A[K,M]^T @ B[K,N]
// Target must be zeroed first. Used for adj^T and br^T reductions (tiny FLOPs).
// ---------------------------------------------------------------------------
__global__ void gemm_tn_atomic(const float* __restrict__ A, const float* __restrict__ B,
                               float* __restrict__ C, int M, int N, int K,
                               int ldc, int coff, int kChunk)
{
    int n = blockIdx.x * blockDim.x + threadIdx.x;
    int m = blockIdx.y;
    if (n >= N) return;
    int kBeg = blockIdx.z * kChunk;
    int kEnd = kBeg + kChunk; if (kEnd > K) kEnd = K;
    float acc = 0.0f;
    for (int k = kBeg; k < kEnd; ++k)
        acc += A[(size_t)k * M + m] * B[(size_t)k * N + n];
    atomicAdd(&C[(size_t)m * ldc + coff + n], acc);
}

// ---------------------------------------------------------------------------
// Row softmax over C cols, zero-pad to W cols.
// ---------------------------------------------------------------------------
__global__ void softmax_pad(const float* __restrict__ in, float* __restrict__ out,
                            int rows, int C, int W)
{
    int r = blockIdx.x * blockDim.x + threadIdx.x;
    if (r >= rows) return;
    const float* x = in + (size_t)r * C;
    float* y = out + (size_t)r * W;
    float mx = -3.4e38f;
    for (int c = 0; c < C; ++c) mx = fmaxf(mx, x[c]);
    float s = 0.0f;
    for (int c = 0; c < C; ++c) { float e = __expf(x[c] - mx); y[c] = e; s += e; }
    float inv = 1.0f / s;
    for (int c = 0; c < C; ++c) y[c] *= inv;
    for (int c = C; c < W; ++c) y[c] = 0.0f;
}

// dst[i, coff:coff+cols] = src[rel[2i+which], 0:cols]
__global__ void gather_rows(const float* __restrict__ src, int scols,
                            const int* __restrict__ rel, int which,
                            float* __restrict__ dst, int ldd, int coff, int nRows)
{
    size_t t = (size_t)blockIdx.x * blockDim.x + threadIdx.x;
    size_t total = (size_t)nRows * scols;
    if (t >= total) return;
    int i = (int)(t / scols), c = (int)(t % scols);
    int s = rel[2 * i + which];
    dst[(size_t)i * ldd + coff + c] = src[(size_t)s * scols + c];
}

// segment_sum: dst[rel[2i+which], coff+c] += src[i, c]
__global__ void segsum_rows(const float* __restrict__ src, int scols,
                            const int* __restrict__ rel, int which,
                            float* __restrict__ dst, int ldd, int coff, int nRows)
{
    size_t t = (size_t)blockIdx.x * blockDim.x + threadIdx.x;
    size_t total = (size_t)nRows * scols;
    if (t >= total) return;
    int i = (int)(t / scols), c = (int)(t % scols);
    int s = rel[2 * i + which];
    atomicAdd(&dst[(size_t)s * ldd + coff + c], src[(size_t)i * scols + c]);
}

// GRU pointwise: z = sigmoid(zpre); rbuf = sigmoid(rpre) * node
__global__ void gru_zr(float* __restrict__ z, float* __restrict__ r,
                       const float* __restrict__ node, size_t n)
{
    size_t t = (size_t)blockIdx.x * blockDim.x + threadIdx.x;
    if (t >= n) return;
    float zv = 1.0f / (1.0f + __expf(-z[t]));
    float rv = 1.0f / (1.0f + __expf(-r[t]));
    z[t] = zv;
    r[t] = rv * node[t];
}

// node = (1-z)*node + z*tanh(h)
__global__ void gru_upd(float* __restrict__ node, const float* __restrict__ z,
                        const float* __restrict__ h, size_t n)
{
    size_t t = (size_t)blockIdx.x * blockDim.x + threadIdx.x;
    if (t >= n) return;
    float zv = z[t];
    node[t] = (1.0f - zv) * node[t] + zv * tanhf(h[t]);
}

// out[c, r] = in[r, c]   (in: R x Ccols; out row stride ldo)
__global__ void transpose_naive(const float* __restrict__ in, float* __restrict__ out,
                                int R, int Ccols, int ldo)
{
    int t = blockIdx.x * blockDim.x + threadIdx.x;
    if (t >= R * Ccols) return;
    int r = t / Ccols, c = t % Ccols;
    out[(size_t)c * ldo + r] = in[(size_t)r * Ccols + c];
}

// ---------------------------------------------------------------------------
// Host orchestration
// ---------------------------------------------------------------------------
extern "C" void kernel_launch(void* const* d_in, const int* in_sizes, int n_in,
                              void* d_out, int out_size, void* d_ws, size_t ws_size,
                              hipStream_t stream)
{
    (void)in_sizes; (void)n_in; (void)out_size; (void)ws_size;

    const int*   rel         = (const int*)d_in[0];
    const float* ent_feats   = (const float*)d_in[1];
    const float* pred_feats  = (const float*)d_in[2];
    const float* ent_logits  = (const float*)d_in[3];
    const float* pred_logits = (const float*)d_in[4];
    const float* emb_ent     = (const float*)d_in[5];
    const float* emb_pred    = (const float*)d_in[6];
    const float* adj_e2e     = (const float*)d_in[7];
    const float* adj_e2p     = (const float*)d_in[8];
    const float* adj_p2e     = (const float*)d_in[9];
    const float* adj_p2p     = (const float*)d_in[10];

    auto Wp = [&](int s) { return (const float*)d_in[11 + 2 * s]; };
    auto Bp = [&](int s) { return (const float*)d_in[12 + 2 * s]; };

    constexpr int NE = 163, NP = 68, NIE = 1024, NIP = 32768;
    constexpr int NEp = 256, NPp = 128;   // padded row capacities (multiple of 128)
    constexpr int H = 512, D2 = 256, D4 = 128, EMB = 300;
    constexpr int W_OE = 7 * D4, W_OP = 8 * D4, W_IE = 3 * D4, W_IP = 3 * D4;

    float* ws = (float*)d_ws;
    size_t off = 0;
    auto alloc = [&](size_t n) {
        float* p = ws + off;
        off += (n + 3) & ~(size_t)3;   // keep every buffer 16B-aligned
        return p;
    };

    float* br_ent  = alloc((size_t)NIE * NE);
    float* br_pred = alloc((size_t)NIP * NP);
    // buffers used as GEMM-A with ragged M are allocated with padded rows
    float* n_oe = alloc((size_t)NEp * H);
    float* n_op = alloc((size_t)NPp * H);
    float* n_ie = alloc((size_t)NIE * H);
    float* n_ip = alloc((size_t)NIP * H);
    float* h_oe = alloc((size_t)NEp * D2);
    float* h_op = alloc((size_t)NPp * D2);
    float* h_ie = alloc((size_t)NIE * D2);
    float* h_ip = alloc((size_t)NIP * D2);
    float* m_oe = alloc((size_t)NE * D4);
    float* m_op = alloc((size_t)NP * D4);
    float* m_ie = alloc((size_t)NIE * D4);
    float* m_ip = alloc((size_t)NIP * D4);
    float* r_oe = alloc((size_t)NEp * W_OE);
    float* r_op = alloc((size_t)NPp * W_OP);
    float* r_ie = alloc((size_t)NIE * W_IE);
    float* r_ip = alloc((size_t)NIP * W_IP);
    float* g_oe = alloc((size_t)NEp * W_OE);
    float* g_op = alloc((size_t)NPp * W_OP);
    float* g_ie = alloc((size_t)NIE * W_IE);
    float* g_ip = alloc((size_t)NIP * W_IP);
    float* u_oe = alloc((size_t)NEp * H);
    float* u_op = alloc((size_t)NPp * H);
    float* u_ie = alloc((size_t)NIE * H);
    float* u_ip = alloc((size_t)NIP * H);
    float* s1 = alloc((size_t)NIP * H);
    float* s2 = alloc((size_t)NIP * H);
    float* s3 = alloc((size_t)NIP * H);
    float* o_oph = alloc((size_t)NPp * H);
    float* o_op  = alloc((size_t)NPp * H);
    float* o_opT = alloc((size_t)H * NPp);   // 512 x 128 (cols 68..127 are pad)

    // full-control launcher: Mpad = readable A rows; ldb/Npad = B stride/capacity
    auto gemmx = [&](const float* A, const float* B, const float* bias, float* C,
                     int M, int Nc, int K, int Mpad, int ldb, int Npad,
                     int ldc, int coff, int flags) {
        dim3 g((Nc + TILE_N - 1) / TILE_N, (M + TILE_M - 1) / TILE_M);
        gemm_nn_bf16<<<g, 256, 0, stream>>>(A, B, bias, C, M, Nc, K,
                                            Mpad, ldb, Npad, ldc, coff, flags);
    };
    auto gemm = [&](const float* A, const float* B, const float* bias, float* C,
                    int M, int Nc, int K, int ldc, int coff, int flags, int Mpad) {
        gemmx(A, B, bias, C, M, Nc, K, Mpad, Nc, Nc, ldc, coff, flags);
    };
    auto tn = [&](const float* A, const float* B, float* C,
                  int M, int Nc, int K, int ldc, int coff, int kchunk) {
        dim3 g((Nc + 127) / 128, M, (K + kchunk - 1) / kchunk);
        gemm_tn_atomic<<<g, 128, 0, stream>>>(A, B, C, M, Nc, K, ldc, coff, kchunk);
    };
    auto ew = [&](size_t n) { return dim3((unsigned)((n + 255) / 256)); };

    // node state copies (never mutate inputs)
    hipMemcpyAsync(n_ie, ent_feats,  sizeof(float) * (size_t)NIE * H, hipMemcpyDeviceToDevice, stream);
    hipMemcpyAsync(n_ip, pred_feats, sizeof(float) * (size_t)NIP * H, hipMemcpyDeviceToDevice, stream);

    // bridge matrices
    softmax_pad<<<ew(NIE), 256, 0, stream>>>(ent_logits,  br_ent,  NIE, 151, NE);
    softmax_pad<<<ew(NIP), 256, 0, stream>>>(pred_logits, br_pred, NIP, 51, NP);

    // init ontology nodes (A is an external input -> Mpad = M)
    gemm(emb_ent,  Wp(0), Bp(0), n_oe, NE, H, EMB, H, 0, 0, NE);
    gemm(emb_pred, Wp(1), Bp(1), n_op, NP, H, EMB, H, 0, 0, NP);

    for (int t = 0; t < 3; ++t) {
        // ---- send MLPs (relu both layers) ----
        gemm(n_oe, Wp(2), Bp(2), h_oe, NE,  D2, H,  D2, 0, FLAG_RELU, NEp);
        gemm(h_oe, Wp(3), Bp(3), m_oe, NE,  D4, D2, D4, 0, FLAG_RELU, NEp);
        gemm(n_op, Wp(4), Bp(4), h_op, NP,  D2, H,  D2, 0, FLAG_RELU, NPp);
        gemm(h_op, Wp(5), Bp(5), m_op, NP,  D4, D2, D4, 0, FLAG_RELU, NPp);
        gemm(n_ie, Wp(6), Bp(6), h_ie, NIE, D2, H,  D2, 0, FLAG_RELU, NIE);
        gemm(h_ie, Wp(7), Bp(7), m_ie, NIE, D4, D2, D4, 0, FLAG_RELU, NIE);
        gemm(n_ip, Wp(8), Bp(8), h_ip, NIP, D2, H,  D2, 0, FLAG_RELU, NIP);
        gemm(h_ip, Wp(9), Bp(9), m_ip, NIP, D4, D2, D4, 0, FLAG_RELU, NIP);

        // ---- zero receive buffers that are accumulated into ----
        hipMemsetAsync(r_oe, 0, sizeof(float) * (size_t)NEp * W_OE, stream);
        hipMemsetAsync(r_op, 0, sizeof(float) * (size_t)NPp * W_OP, stream);
        hipMemsetAsync(r_ie, 0, sizeof(float) * (size_t)NIE * W_IE, stream);

        // ---- r_oe = [adj_e2e^T m_oe | adj_p2e^T m_op | br_ent^T m_ie] ----
        for (int e = 0; e < 3; ++e)
            tn(adj_e2e + (size_t)e * NE * NE, m_oe, r_oe, NE, D4, NE, W_OE, e * D4, NE);
        for (int e = 0; e < 3; ++e)
            tn(adj_p2e + (size_t)e * NP * NE, m_op, r_oe, NE, D4, NP, W_OE, 3 * D4 + e * D4, NP);
        tn(br_ent, m_ie, r_oe, NE, D4, NIE, W_OE, 6 * D4, 256);

        // ---- r_op = [adj_e2p^T m_oe | adj_p2p^T m_op | br_pred^T m_ip] ----
        for (int e = 0; e < 3; ++e)
            tn(adj_e2p + (size_t)e * NE * NP, m_oe, r_op, NP, D4, NE, W_OP, e * D4, NE);
        for (int e = 0; e < 4; ++e)
            tn(adj_p2p + (size_t)e * NP * NP, m_op, r_op, NP, D4, NP, W_OP, 3 * D4 + e * D4, NP);
        tn(br_pred, m_ip, r_op, NP, D4, NIP, W_OP, 7 * D4, 2048);

        // ---- r_ie = [segsum_subj m_ip | segsum_obj m_ip | br_ent m_oe] ----
        segsum_rows<<<ew((size_t)NIP * D4), 256, 0, stream>>>(m_ip, D4, rel, 0, r_ie, W_IE, 0,  NIP);
        segsum_rows<<<ew((size_t)NIP * D4), 256, 0, stream>>>(m_ip, D4, rel, 1, r_ie, W_IE, D4, NIP);
        gemm(br_ent, m_oe, nullptr, r_ie, NIE, D4, NE, W_IE, 2 * D4, 0, NIE);

        // ---- r_ip = [m_ie[subj] | m_ie[obj] | br_pred m_op] ----
        gather_rows<<<ew((size_t)NIP * D4), 256, 0, stream>>>(m_ie, D4, rel, 0, r_ip, W_IP, 0,  NIP);
        gather_rows<<<ew((size_t)NIP * D4), 256, 0, stream>>>(m_ie, D4, rel, 1, r_ip, W_IP, D4, NIP);
        gemm(br_pred, m_op, nullptr, r_ip, NIP, D4, NP, W_IP, 2 * D4, 0, NIP);

        // ---- recv MLPs (relu both layers) ----
        gemm(r_oe, Wp(10), Bp(10), g_oe, NE,  W_OE, W_OE, W_OE, 0, FLAG_RELU, NEp);
        gemm(g_oe, Wp(11), Bp(11), u_oe, NE,  H,    W_OE, H,    0, FLAG_RELU, NEp);
        gemm(r_op, Wp(12), Bp(12), g_op, NP,  W_OP, W_OP, W_OP, 0, FLAG_RELU, NPp);
        gemm(g_op, Wp(13), Bp(13), u_op, NP,  H,    W_OP, H,    0, FLAG_RELU, NPp);
        gemm(r_ie, Wp(14), Bp(14), g_ie, NIE, W_IE, W_IE, W_IE, 0, FLAG_RELU, NIE);
        gemm(g_ie, Wp(15), Bp(15), u_ie, NIE, H,    W_IE, H,    0, FLAG_RELU, NIE);
        gemm(r_ip, Wp(16), Bp(16), g_ip, NIP, W_IP, W_IP, W_IP, 0, FLAG_RELU, NIP);
        gemm(g_ip, Wp(17), Bp(17), u_ip, NIP, H,    W_IP, H,    0, FLAG_RELU, NIP);

        // ---- GRUs ----
        struct GruDesc { float* node; const float* msg; int M; int Mp; int base; };
        GruDesc G[4] = { {n_oe, u_oe, NE, NEp, 18}, {n_op, u_op, NP, NPp, 24},
                         {n_ie, u_ie, NIE, NIE, 30}, {n_ip, u_ip, NIP, NIP, 36} };
        for (int gi = 0; gi < 4; ++gi) {
            float* node = G[gi].node; const float* msg = G[gi].msg;
            int M = G[gi].M, Mp = G[gi].Mp, b = G[gi].base;
            size_t n = (size_t)M * H;
            // z_pre = msg@eq3w + b3w + node@eq3u + b3u
            gemm(msg,  Wp(b + 0), Bp(b + 0), s1, M, H, H, H, 0, 0,        Mp);
            gemm(node, Wp(b + 1), Bp(b + 1), s1, M, H, H, H, 0, FLAG_ACC, Mp);
            // r_pre = msg@eq4w + b4w + node@eq4u + b4u
            gemm(msg,  Wp(b + 2), Bp(b + 2), s2, M, H, H, H, 0, 0,        Mp);
            gemm(node, Wp(b + 3), Bp(b + 3), s2, M, H, H, H, 0, FLAG_ACC, Mp);
            gru_zr<<<ew(n), 256, 0, stream>>>(s1, s2, node, n);  // s1=z, s2=r*node
            // h_pre = msg@eq5w + b5w + (r*node)@eq5u + b5u
            gemm(msg, Wp(b + 4), Bp(b + 4), s3, M, H, H, H, 0, 0,        Mp);
            gemm(s2,  Wp(b + 5), Bp(b + 5), s3, M, H, H, H, 0, FLAG_ACC,
                 (Mp > NIP ? Mp : NIP));   // s2 has NIP-row capacity
            gru_upd<<<ew(n), 256, 0, stream>>>(node, s1, s3, n);
        }
    }

    // ---- output head ----
    gemm(n_ip, Wp(42), Bp(42), s1, NIP, H, H, H, 0, FLAG_RELU, NIP);
    gemm(s1,   Wp(43), Bp(43), s2, NIP, H, H, H, 0, 0,         NIP);   // o_ip
    gemm(n_op, Wp(44), Bp(44), o_oph, NP, H, H, H, 0, FLAG_RELU, NPp);
    gemm(o_oph, Wp(45), Bp(45), o_op, NP, H, H, H, 0, 0,         NPp); // o_op
    transpose_naive<<<ew((size_t)NP * H), 256, 0, stream>>>(o_op, o_opT, NP, H, NPp);

    // out = o_ip @ o_op^T  -> (32768 x 68); B has padded 128-col stride -> fast path
    gemmx(s2, o_opT, nullptr, (float*)d_out, NIP, NP, H, NIP, NPp, NPp, NP, 0, 0);
}
